// LazyCrossAttentionGQASDPA_28424093565386
// MI455X (gfx1250) — matrix operand order repaired and checked by reading, not compiled, not run on hardware
//
#include <hip/hip_runtime.h>
#include <hip/hip_bf16.h>
#include <cstddef>

// ---------------------------------------------------------------------------
// Problem constants (B,TQ,TK,D,H,G = 2,2048,2048,2048,32,8; DH=64, REP=4)
// ---------------------------------------------------------------------------
#define BATCH   2
#define TQ      2048
#define TKK     2048
#define DMODEL  2048
#define NHEAD   32
#define NGRP    8
#define DHEAD   64
#define EPSV    1e-6f
#define NEGINF  (-10000.0f)
#define SCALE   0.125f   // 1/sqrt(64), folded into Q projection store

typedef __attribute__((ext_vector_type(8)))  _Float16 h8;
typedef __attribute__((ext_vector_type(16))) _Float16 h16;
typedef __attribute__((ext_vector_type(8)))  float    f8;

// ---------------------------------------------------------------------------
// WMMA wrapper: D = A(16x32 f16) * B(32x16 f16) + C(16x16 f32)
// ---------------------------------------------------------------------------
__device__ __forceinline__ f8 wmma_f16(h16 a, h16 b, f8 c) {
  return __builtin_amdgcn_wmma_f32_16x16x32_f16(
      /*neg_a=*/false, a, /*neg_b=*/false, b,
      /*c_mod=*/(short)0, c, /*reuse_a=*/false, /*reuse_b=*/false);
}

// ds_swizzle XOR within 16-lane halves (group-of-32 mode: and=0x1f, or=0, xor=XM)
template <int XM>
__device__ __forceinline__ float swz_xor(float v) {
  return __int_as_float(
      __builtin_amdgcn_ds_swizzle(__float_as_int(v), (XM << 10) | 0x1f));
}
__device__ __forceinline__ float red_max16(float t) {
  t = fmaxf(t, swz_xor<1>(t));
  t = fmaxf(t, swz_xor<2>(t));
  t = fmaxf(t, swz_xor<4>(t));
  t = fmaxf(t, swz_xor<8>(t));
  return t;
}
__device__ __forceinline__ float red_sum16(float t) {
  t += swz_xor<1>(t);
  t += swz_xor<2>(t);
  t += swz_xor<4>(t);
  t += swz_xor<8>(t);
  return t;
}

// A fragment (16x32, MxK), row-major source, per ISA 7.12.2:
// lane l: row m = l&15; K-base kb = (l>=16 ? 8 : 0);
// VGPR0-3 hold K=kb..kb+7, VGPR4-7 hold K=kb+16..kb+23  -> two 16B loads.
__device__ __forceinline__ h16 load_frag_a(const _Float16* __restrict__ base,
                                           int ld, int lane) {
  const int m  = lane & 15;
  const int kb = (lane >> 4) << 3;
  const _Float16* p = base + (size_t)m * ld + kb;
  h8 lo = *reinterpret_cast<const h8*>(p);
  h8 hi = *reinterpret_cast<const h8*>(p + 16);
  return __builtin_shufflevector(lo, hi, 0,1,2,3,4,5,6,7,8,9,10,11,12,13,14,15);
}

// B fragment (32x16, KxN) where B[k][n] = SRC[n][k] (row-major SRC, N rows of K):
// lane l: col n = l&15; kb = (l>=16 ? 8 : 0); elems 0-7 hold K=kb..kb+7,
// elems 8-15 hold K=kb+16..kb+23 -> two 16B loads per lane.
__device__ __forceinline__ h16 load_frag_b(const _Float16* __restrict__ base,
                                           int ld, int lane) {
  const int n  = lane & 15;
  const int kc = (lane >> 4) << 3;
  const _Float16* p = base + (size_t)n * ld + kc;
  h8 lo = *reinterpret_cast<const h8*>(p);
  h8 hi = *reinterpret_cast<const h8*>(p + 16);
  return __builtin_shufflevector(lo, hi, 0,1,2,3,4,5,6,7,8,9,10,11,12,13,14,15);
}

// ---------------------------------------------------------------------------
// Kernel 1: fused RMSNorm + cast to f16.  One block per row of x (B*TQ rows).
// ---------------------------------------------------------------------------
__global__ __launch_bounds__(256) void rms_cast_kernel(
    const float* __restrict__ x, const float* __restrict__ norm_w,
    _Float16* __restrict__ xn) {
  const int row = blockIdx.x;
  const float* xr = x + (size_t)row * DMODEL;
  __shared__ float red[256];
  float vals[8];
  float s = 0.f;
#pragma unroll
  for (int j = 0; j < 8; ++j) {
    vals[j] = xr[threadIdx.x + j * 256];
    s += vals[j] * vals[j];
  }
  red[threadIdx.x] = s;
  __syncthreads();
  for (int off = 128; off > 0; off >>= 1) {
    if (threadIdx.x < off) red[threadIdx.x] += red[threadIdx.x + off];
    __syncthreads();
  }
  const float scale = rsqrtf(red[0] * (1.0f / DMODEL) + EPSV);
#pragma unroll
  for (int j = 0; j < 8; ++j) {
    const int c = threadIdx.x + j * 256;
    xn[(size_t)row * DMODEL + c] = (_Float16)(vals[j] * scale * norm_w[c]);
  }
}

// ---------------------------------------------------------------------------
// Kernel 2: K -> f16 (same layout), V -> f16 transposed [b,g,dh,tk],
//           mask(bool) -> float additive mask (0 / -10000).
// ---------------------------------------------------------------------------
__global__ __launch_bounds__(256) void kv_prep_kernel(
    const float* __restrict__ k_ctx, const float* __restrict__ v_ctx,
    const unsigned char* __restrict__ mask,
    _Float16* __restrict__ Kf, _Float16* __restrict__ Vt,
    float* __restrict__ maskf) {
  const size_t idx = (size_t)blockIdx.x * 256 + threadIdx.x; // B*G*TK*DH = 2^21
  const int dh = idx & 63;
  const int tk = (idx >> 6) & (TKK - 1);
  const int bg = (int)(idx >> 17);
  Kf[idx] = (_Float16)k_ctx[idx];
  Vt[((size_t)bg * DHEAD + dh) * TKK + tk] = (_Float16)v_ctx[idx];
  if (idx < (size_t)BATCH * TKK) maskf[idx] = mask[idx] ? 0.0f : NEGINF;
}

// ---------------------------------------------------------------------------
// Kernel 3: weights -> f16
// ---------------------------------------------------------------------------
__global__ __launch_bounds__(256) void wcast_kernel(
    const float* __restrict__ Wq, const float* __restrict__ Wout,
    _Float16* __restrict__ Wqh, _Float16* __restrict__ Wouth) {
  const size_t i = (size_t)blockIdx.x * 256 + threadIdx.x;
  Wqh[i]   = (_Float16)Wq[i];
  Wouth[i] = (_Float16)Wout[i];
}

// ---------------------------------------------------------------------------
// WMMA GEMM:  C[M,N] = A[M,K] @ W[N,K]^T   (A,W f16 row-major)
// MODE 0: store f16*SCALE into Q layout [B,H,TQ,DH] (head-split remap)
// MODE 1: store f32 row-major into out[M,N]
// Block: 256 thr = 8 waves (4 in M x 2 in N), wave tile 32x64, block 128x128.
// 8 WMMA per 6 fragment loads per k-step.
// ---------------------------------------------------------------------------
template <int MODE>
__global__ __launch_bounds__(256) void gemm_f16_kernel(
    const _Float16* __restrict__ A, const _Float16* __restrict__ W,
    void* __restrict__ out, int K) {
  const int lane = threadIdx.x & 31;
  const int wave = threadIdx.x >> 5;
  const int wm = wave >> 1;       // 0..3
  const int wn = wave & 1;        // 0..1
  const int m0 = blockIdx.y * 128 + wm * 32;
  const int n0 = blockIdx.x * 128 + wn * 64;

  const _Float16* Ab = A + (size_t)m0 * K;
  const _Float16* Wb = W + (size_t)n0 * K;

  f8 acc[2][4] = {};
  for (int k0 = 0; k0 < K; k0 += 32) {
    __builtin_prefetch(Ab + k0 + 256, 0, 3);   // global_prefetch_b8
    __builtin_prefetch(Wb + k0 + 256, 0, 3);
    h16 a[2], b[4];
#pragma unroll
    for (int mi = 0; mi < 2; ++mi)
      a[mi] = load_frag_a(Ab + (size_t)(mi * 16) * K + k0, K, lane);
#pragma unroll
    for (int nj = 0; nj < 4; ++nj)
      b[nj] = load_frag_b(Wb + (size_t)(nj * 16) * K + k0, K, lane);
#pragma unroll
    for (int mi = 0; mi < 2; ++mi)
#pragma unroll
      for (int nj = 0; nj < 4; ++nj)
        acc[mi][nj] = wmma_f16(a[mi], b[nj], acc[mi][nj]);
  }

  // C layout: VGPR i, lanes 0-15 -> row i, lanes 16-31 -> row i+8; col = lane&15
  const int h2 = lane >> 4;
  const int cc = lane & 15;
#pragma unroll
  for (int mi = 0; mi < 2; ++mi) {
#pragma unroll
    for (int nj = 0; nj < 4; ++nj) {
#pragma unroll
      for (int i = 0; i < 8; ++i) {
        const int r = m0 + mi * 16 + i + 8 * h2;
        const int c = n0 + nj * 16 + cc;
        if (MODE == 0) {
          // remap [B*TQ, D] -> Q f16 [B,H,TQ,DH], pre-scaled by 1/sqrt(DH)
          const int b_ = r >> 11, tq = r & (TQ - 1);
          const int h = c >> 6, dh = c & 63;
          ((_Float16*)out)[(((size_t)(b_ * NHEAD + h)) * TQ + tq) * DHEAD + dh] =
              (_Float16)(acc[mi][nj][i] * SCALE);
        } else {
          ((float*)out)[(size_t)r * DMODEL + c] = acc[mi][nj][i];
        }
      }
    }
  }
}

// ---------------------------------------------------------------------------
// Flash-attention (GQA).  One wave handles 16 Q rows of one head, full TK
// sweep in 64-key chunks with online softmax.  Block = 8 waves; 2KB private
// LDS slab per wave for the C-layout -> A-layout transpose of P.
//   Q  : [B,H,TQ,DH] f16 (pre-scaled by 1/sqrt(DH))
//   K  : [B,G,TK,DH] f16
//   Vt : [B,G,DH,TK] f16
//   Y  : [B*TQ, D]   f16 (merged heads)
// ---------------------------------------------------------------------------
__global__ __launch_bounds__(256) void attn_kernel(
    const _Float16* __restrict__ Q, const _Float16* __restrict__ Kf,
    const _Float16* __restrict__ Vt, const float* __restrict__ maskf,
    _Float16* __restrict__ Y) {
  __shared__ _Float16 ldsP[8 * 16 * 64];   // 16 KB

  const int lane = threadIdx.x & 31;
  const int wave = threadIdx.x >> 5;
  const int unit = blockIdx.x * 8 + wave;  // 0 .. B*H*(TQ/16)-1 = 8191
  const int qb   = unit & 127;             // q tile
  const int head = (unit >> 7) & (NHEAD - 1);
  const int b    = unit >> 12;
  const int g    = head >> 2;              // REP = 4
  const int q0   = qb * 16;

  const _Float16* Qb = Q  + (((size_t)(b * NHEAD + head)) * TQ + q0) * DHEAD;
  const _Float16* Kb = Kf + ((size_t)(b * NGRP + g)) * TKK * DHEAD;
  const _Float16* Vb = Vt + ((size_t)(b * NGRP + g)) * DHEAD * TKK;
  const float*    Mb = maskf + (size_t)b * TKK;
  _Float16* Pl = ldsP + wave * (16 * 64);

  // Q fragments (A layout), reused for the whole K sweep
  const h16 qa0 = load_frag_a(Qb, DHEAD, lane);       // dh 0..31
  const h16 qa1 = load_frag_a(Qb + 32, DHEAD, lane);  // dh 32..63

  f8 o[4] = {};                 // O tile: 16 x 64 (4 x 16-col groups)
  float mrow[8], lrow[8];
#pragma unroll
  for (int i = 0; i < 8; ++i) { mrow[i] = -INFINITY; lrow[i] = 0.f; }

  const int h2 = lane >> 4;
  const int cc = lane & 15;

  for (int tk0 = 0; tk0 < TKK; tk0 += 64) {
    __builtin_prefetch(Kb + (size_t)(tk0 + 64) * DHEAD, 0, 3);
    // ---- logits S (16 x 64) via (Q*scale) K^T --------------------------
    f8 s[4] = {};
#pragma unroll
    for (int t = 0; t < 4; ++t) {
      const _Float16* kb = Kb + (size_t)(tk0 + 16 * t) * DHEAD;
      s[t] = wmma_f16(qa0, load_frag_b(kb,      DHEAD, lane), s[t]);
      s[t] = wmma_f16(qa1, load_frag_b(kb + 32, DHEAD, lane), s[t]);
    }
    float madd[4];
#pragma unroll
    for (int t = 0; t < 4; ++t) madd[t] = Mb[tk0 + 16 * t + cc];

    // ---- online softmax update ----------------------------------------
#pragma unroll
    for (int i = 0; i < 8; ++i) {
      float v[4];
#pragma unroll
      for (int t = 0; t < 4; ++t) v[t] = s[t][i] + madd[t];
      // row max across 16 columns (ds_swizzle XOR stays within 16-lane half)
      const float tmax = red_max16(fmaxf(fmaxf(v[0], v[1]), fmaxf(v[2], v[3])));
      const float mnew  = fmaxf(mrow[i], tmax);
      const float alpha = __expf(mrow[i] - mnew);
      mrow[i] = mnew;
      float rs = 0.f;
      const int rbase = (i + 8 * h2) * 64 + cc;
#pragma unroll
      for (int t = 0; t < 4; ++t) {
        const float p = __expf(v[t] - mnew);
        rs += p;
        Pl[rbase + 16 * t] = (_Float16)p;   // spill P (C layout) to LDS
      }
      lrow[i] = lrow[i] * alpha + red_sum16(rs);
#pragma unroll
      for (int j = 0; j < 4; ++j) o[j][i] *= alpha;   // rescale O rows
    }

    // ---- P (A layout) from LDS, then O += P @ V ------------------------
    const h16 pa0 = load_frag_a(Pl,      64, lane);  // keys tk0..tk0+31
    const h16 pa1 = load_frag_a(Pl + 32, 64, lane);  // keys tk0+32..tk0+63
#pragma unroll
    for (int j = 0; j < 4; ++j) {
      const _Float16* vb = Vb + (size_t)(j * 16) * TKK + tk0;
      o[j] = wmma_f16(pa0, load_frag_b(vb,      TKK, lane), o[j]);
      o[j] = wmma_f16(pa1, load_frag_b(vb + 32, TKK, lane), o[j]);
    }
  }

  // ---- normalize and store merged-head Y -------------------------------
#pragma unroll
  for (int i = 0; i < 8; ++i) {
    const float inv = 1.0f / lrow[i];
    const int r = q0 + i + 8 * h2;
    const size_t rowbase = ((size_t)b * TQ + r) * DMODEL + head * DHEAD;
#pragma unroll
    for (int j = 0; j < 4; ++j)
      Y[rowbase + j * 16 + cc] = (_Float16)(o[j][i] * inv);
  }
}

// ---------------------------------------------------------------------------
// Host launcher
// ---------------------------------------------------------------------------
extern "C" void kernel_launch(void* const* d_in, const int* in_sizes, int n_in,
                              void* d_out, int out_size, void* d_ws, size_t ws_size,
                              hipStream_t stream) {
  (void)in_sizes; (void)n_in; (void)out_size; (void)ws_size;
  const float*         x    = (const float*)d_in[0];
  const float*         kc   = (const float*)d_in[1];
  const float*         vc   = (const float*)d_in[2];
  const unsigned char* mask = (const unsigned char*)d_in[3];
  const float*         Wq   = (const float*)d_in[4];
  const float*         Wout = (const float*)d_in[5];
  const float*         nw   = (const float*)d_in[6];
  float* out = (float*)d_out;

  char* ws = (char*)d_ws;                       // offsets (bytes)
  _Float16* xn    = (_Float16*)(ws + 0);        // 16 MiB  B*TQ*D f16
  _Float16* Qf    = (_Float16*)(ws + 16777216); // 16 MiB  [B,H,TQ,DH]
  _Float16* Kf    = (_Float16*)(ws + 33554432); //  4 MiB  [B,G,TK,DH]
  _Float16* Vt    = (_Float16*)(ws + 37748736); //  4 MiB  [B,G,DH,TK]
  _Float16* Wqh   = (_Float16*)(ws + 41943040); //  8 MiB
  _Float16* Wouth = (_Float16*)(ws + 50331648); //  8 MiB
  _Float16* Y     = (_Float16*)(ws + 58720256); // 16 MiB  [B*TQ, D]
  float*    maskf = (float*)   (ws + 75497472); // 16 KiB

  // 1) RMSNorm + f16 cast (one block per row)
  rms_cast_kernel<<<BATCH * TQ, 256, 0, stream>>>(x, nw, xn);
  // 2) K/V prep (+ mask floatize)
  kv_prep_kernel<<<(BATCH * NGRP * TKK * DHEAD) / 256, 256, 0, stream>>>(
      kc, vc, mask, Kf, Vt, maskf);
  // 3) weights -> f16
  wcast_kernel<<<(DMODEL * DMODEL) / 256, 256, 0, stream>>>(Wq, Wout, Wqh, Wouth);
  // 4) Q projection: [4096,2048] = xn @ Wq^T -> Q f16 head-split, pre-scaled
  gemm_f16_kernel<0><<<dim3(DMODEL / 128, (BATCH * TQ) / 128), 256, 0, stream>>>(
      xn, Wqh, (void*)Qf, DMODEL);
  // 5) GQA flash attention -> merged-head Y f16
  attn_kernel<<<(BATCH * NHEAD * (TQ / 16)) / 8, 256, 0, stream>>>(
      Qf, Kf, Vt, maskf, Y);
  // 6) Output projection: out = Y @ Wout^T (f32)
  gemm_f16_kernel<1><<<dim3(DMODEL / 128, (BATCH * TQ) / 128), 256, 0, stream>>>(
      Y, Wouth, (void*)out, DMODEL);
}
